// ComputeTotalSim_84267258347855
// MI455X (gfx1250) — matrix operand order repaired and checked by reading, not compiled, not run on hardware
//
#include <hip/hip_runtime.h>
#include <hip/hip_bf16.h>

// Problem dims (from setup_inputs): features [16, 512, 56, 56] f32, k = 3.
#define HH 56
#define WW 56
#define CTOT 512
#define NB 16
#define CC 32                          // channels per LDS chunk
#define NCHUNK (CTOT / CC)             // 16
#define TPX 16                         // pixel-tile width  (one wave row = 16 px -> WMMA M/N)
#define TPY 8                          // pixel-tile height (8 waves per block)
#define HALO_W 20                      // TPX + 2*2 halo
#define HALO_H 12                      // TPY + 2*2 halo
#define PLANE (HALO_H * HALO_W)        // 240 floats per channel plane
#define TILEF (CC * PLANE)             // 7680 floats per tile buffer
#define S1W (CC * HALO_W)              // 640 floats per sum-variant per wave
#define S1F (3 * S1W)                  // 1920 floats col-sum staging per wave
#define G16H (4 * 16 * CC)             // 2048 halves f16 fragment staging per wave
#define SMEM_BYTES ((2 * TILEF + 8 * S1F) * 4 + 8 * G16H * 2)  // 155648 B < 320 KB

typedef _Float16 v8h  __attribute__((ext_vector_type(8)));
typedef _Float16 v16h __attribute__((ext_vector_type(16)));
typedef _Float16 h2   __attribute__((ext_vector_type(2)));
typedef float    v8f  __attribute__((ext_vector_type(8)));
typedef unsigned int tdm_u32x4 __attribute__((ext_vector_type(4)));
typedef int          tdm_i32x8 __attribute__((ext_vector_type(8)));
typedef int          tdm_i32x4 __attribute__((ext_vector_type(4)));

#if defined(__HIP_DEVICE_COMPILE__) && defined(__has_builtin)
#if __has_builtin(__builtin_amdgcn_tensor_load_to_lds) && __has_builtin(__builtin_amdgcn_s_wait_tensorcnt)
#define USE_TDM 1
#endif
#endif
#ifndef USE_TDM
#define USE_TDM 0
#endif

__device__ __forceinline__ float pick8(v8f a, int i) {
  float r = a[0];
  r = (i == 1) ? a[1] : r;
  r = (i == 2) ? a[2] : r;
  r = (i == 3) ? a[3] : r;
  r = (i == 4) ? a[4] : r;
  r = (i == 5) ? a[5] : r;
  r = (i == 6) ? a[6] : r;
  r = (i == 7) ? a[7] : r;
  return r;
}

#if USE_TDM
// Build a 3D (x=20, y=12, z=CC) tile descriptor per CDNA5 ISA ch.8 (D# groups)
// and fire the Tensor Data Mover. tensor_dim0/1 are set to the remaining
// W/H extent so right/bottom out-of-range reads are zero-filled by hardware.
// clang-23 (therock lane) exposes the 6-argument builtin:
//   (uint32x4 g0, int32x8 g1, int32x4 g2, int32x4 g3, int32x8 extra, i32 cpol)
__device__ __forceinline__ void tdm_issue(const float* gptr, unsigned lds_byte,
                                          unsigned td0, unsigned td1) {
  unsigned long long ga = (unsigned long long)gptr;
  tdm_u32x4 g0;
  g0[0] = 1u;                                         // count = 1 valid descriptor
  g0[1] = lds_byte;                                   // LDS destination byte addr
  g0[2] = (unsigned)(ga & 0xFFFFFFFFull);             // global_addr[31:0]
  g0[3] = ((unsigned)(ga >> 32) & 0x01FFFFFFu) | (2u << 30);  // addr[56:32] | type=2
  const unsigned long long s0 = (unsigned long long)WW;        // dim0 stride (elems)
  const unsigned long long s1 = (unsigned long long)(HH * WW); // dim1 stride (elems)
  tdm_i32x8 g1;
  g1[0] = (int)(2u << 16);                            // data_size = 4B, no multicast
  g1[1] = (int)((td0 & 0xFFFFu) << 16);               // tensor_dim0[15:0]
  g1[2] = (int)(((td0 >> 16) & 0xFFFFu) | ((td1 & 0xFFFFu) << 16));
  g1[3] = (int)(((td1 >> 16) & 0xFFFFu) | ((unsigned)HALO_W << 16)); // tile_dim0=20
  g1[4] = (int)((unsigned)HALO_H | ((unsigned)CC << 16));            // tile_dim1/2
  g1[5] = (int)(unsigned)(s0 & 0xFFFFFFFFull);
  g1[6] = (int)(((unsigned)((s0 >> 32) & 0xFFFFull)) |
                (((unsigned)(s1 & 0xFFFFull)) << 16));
  g1[7] = (int)(unsigned)(s1 >> 16);
  tdm_i32x4 g2; g2[0] = CC; g2[1] = 0; g2[2] = 0; g2[3] = 0;  // tensor_dim2 = CC
  tdm_i32x4 g3; g3[0] = 0;  g3[1] = 0; g3[2] = 0; g3[3] = 0;
  tdm_i32x8 gx; gx[0] = 0; gx[1] = 0; gx[2] = 0; gx[3] = 0;
  gx[4] = 0; gx[5] = 0; gx[6] = 0; gx[7] = 0;
  __builtin_amdgcn_tensor_load_to_lds(g0, g1, g2, g3, gx, 0);
}
#endif

__global__ __launch_bounds__(256) void ComputeTotalSim_84267258347855_kernel(
    const float* __restrict__ f, float* __restrict__ out) {
  extern __shared__ char smem[];
  float* tiles = (float*)smem;

  const int tid  = (int)threadIdx.x;
  const int w    = tid >> 5;            // wave id = row within pixel tile
  const int lane = tid & 31;
  const int b  = (int)blockIdx.z;
  const int x0 = (int)blockIdx.x * TPX;
  const int y0 = (int)blockIdx.y * TPY;
  const int y  = y0 + w;                // global output row (< 56 always)
  const int sx_base = (x0 - 2 > 0) ? x0 - 2 : 0;   // tile fetch origin (clamped)
  const int sy_base = (y0 - 2 > 0) ? y0 - 2 : 0;
  const unsigned td0 = (unsigned)(WW - sx_base);
  const unsigned td1 = (unsigned)(HH - sy_base);

  float*    s1base = tiles + 2 * TILEF + w * S1F;    // [3][CC][HALO_W] per wave
  _Float16* g16 = (_Float16*)(smem + (2 * TILEF + 8 * S1F) * 4) + w * G16H;

  const float* fb = f + (size_t)b * CTOT * HH * WW;
  const size_t tile_goff = (size_t)sy_base * WW + sx_base;

#if USE_TDM
  const unsigned lds0 = (unsigned)(unsigned long long)(void*)tiles;
#endif

  // Six WMMA f32 accumulators: dotH, dotV, |g_r|^2, |g_l|^2, |g_t|^2, |g_b|^2.
  v8f accH = {}, accV = {}, accNR = {}, accNL = {}, accNT = {}, accNB = {};

  // ---- per-lane geometry (uniform across channel loop) ----
  const int p    = lane & 15;           // pixel within the wave's 16-px row
  const int half = lane >> 4;           // fragment half (ISA A/B lane split)
  const int x    = (x0 + p < WW) ? (x0 + p) : (WW - 1);  // clamp; result discarded
  const int qx   = x - sx_base;
  const bool mL = (x >= 1), mR = (x + 1 < WW);           // output-coord tap masks
  const int jr0 = ((x - 2 > 0) ? x - 2 : 0) - sx_base;   // g_r clamped source cols
  const int jr1 = ((x - 1 > 0) ? x - 1 : 0) - sx_base;
  const int jl1 = ((x + 1 < WW) ? x + 1 : WW - 1) - sx_base;  // g_l clamped cols
  const int jl2 = ((x + 2 < WW) ? x + 2 : WW - 1) - sx_base;
  const int qm1 = (qx - 1 > 0) ? qx - 1 : 0;
  const int qp1 = qx + 1;

  // Row guards / clamped local row indices (uniform per wave).
  const bool gm2 = (y >= 2), gm1 = (y >= 1), gp1 = (y + 1 < HH), gp2 = (y + 2 < HH);
  const int ry  = y - sy_base;                      // local row of y, <= 9
  const int rm2 = (ry - 2 > 0) ? ry - 2 : 0;
  const int rm1 = (ry - 1 > 0) ? ry - 1 : 0;

#if USE_TDM
  if (tid < 32) tdm_issue(fb + tile_goff, lds0, td0, td1);  // prologue: chunk 0
#endif

  for (int ch = 0; ch < NCHUNK; ++ch) {
    float* buf = tiles + (ch & 1) * TILEF;
#if USE_TDM
    if (tid < 32) {
      if (ch + 1 < NCHUNK) {        // prefetch next chunk into other buffer
        tdm_issue(fb + (size_t)(ch + 1) * CC * HH * WW + tile_goff,
                  lds0 + (unsigned)(((ch + 1) & 1) * (TILEF * 4)), td0, td1);
        __builtin_amdgcn_s_wait_tensorcnt(1);   // oldest (current chunk) done
      } else {
        __builtin_amdgcn_s_wait_tensorcnt(0);
      }
    }
#else
    {
      const float* fc = fb + (size_t)ch * CC * HH * WW;
      for (int i = tid; i < TILEF; i += 256) {
        int c = i / PLANE;
        int rem = i - c * PLANE;
        int r = rem / HALO_W;
        int q = rem - r * HALO_W;
        int gy = sy_base + r, gx = sx_base + q;
        float v = 0.f;
        if (gy < HH && gx < WW) v = fc[(size_t)c * (HH * WW) + (size_t)gy * WW + gx];
        buf[i] = v;
      }
    }
#endif
    __syncthreads();

    // ---- stage 1: masked vertical column sums (plain / top-shift / bot-shift)
    //      lane == channel (CC == 32), 20 halo columns each, for this wave's row y.
    {
      const float* plane = buf + lane * PLANE;
      float* sP = s1base + 0 * S1W + lane * HALO_W;
      float* sT = s1base + 1 * S1W + lane * HALO_W;
      float* sB = s1base + 2 * S1W + lane * HALO_W;
      for (int q = 0; q < HALO_W; ++q) {
        float fm2 = gm2 ? plane[rm2 * HALO_W + q] : 0.f;
        float fm1 = gm1 ? plane[rm1 * HALO_W + q] : 0.f;
        float f0  = plane[ry * HALO_W + q];
        float fp1 = gp1 ? plane[(ry + 1) * HALO_W + q] : 0.f;
        float fp2 = gp2 ? plane[(ry + 2) * HALO_W + q] : 0.f;
        sP[q] = fm1 + f0 + fp1;
        sT[q] = (gm1 ? f0 : 0.f) + (gp1 ? fp1 : f0) +
                (gp1 ? (gp2 ? fp2 : fp1) : 0.f);
        sB[q] = (gm1 ? (gm2 ? fm2 : fm1) : 0.f) + (gm1 ? fm1 : f0) +
                (gp1 ? f0 : 0.f);
      }
    }
    __syncthreads();

    // ---- stage 2: horizontal combine -> 4 directional box-filtered values,
    //      converted to f16 and staged in WMMA fragment layout [dir][pixel][K].
    {
      const float inv9 = (1.0f / 9.0f);
#pragma unroll
      for (int jj = 0; jj < 8; ++jj) {
        const int ca = half * 16 + 2 * jj;       // this lane's channel pair
        h2 pk0, pk1, pk2, pk3;
#pragma unroll
        for (int e = 0; e < 2; ++e) {
          const int c = ca + e;
          const float* sP = s1base + 0 * S1W + c * HALO_W;
          const float* sT = s1base + 1 * S1W + c * HALO_W;
          const float* sB = s1base + 2 * S1W + c * HALO_W;
          float gr = (mL ? sP[jr0] : 0.f) + sP[jr1] + (mR ? sP[qx] : 0.f);
          float gl = (mL ? sP[qx] : 0.f) + sP[jl1] + (mR ? sP[jl2] : 0.f);
          float gt = (mL ? sT[qm1] : 0.f) + sT[qx] + (mR ? sT[qp1] : 0.f);
          float gb = (mL ? sB[qm1] : 0.f) + sB[qx] + (mR ? sB[qp1] : 0.f);
          pk0[e] = (_Float16)(gr * inv9);
          pk1[e] = (_Float16)(gl * inv9);
          pk2[e] = (_Float16)(gt * inv9);
          pk3[e] = (_Float16)(gb * inv9);
        }
        *(h2*)(g16 + ((0 * 16 + p) * CC + ca)) = pk0;
        *(h2*)(g16 + ((1 * 16 + p) * CC + ca)) = pk1;
        *(h2*)(g16 + ((2 * 16 + p) * CC + ca)) = pk2;
        *(h2*)(g16 + ((3 * 16 + p) * CC + ca)) = pk3;
      }
    }
    __syncthreads();

    // ---- fragment loads (ISA 7.12.2 16-bit layouts) + 6 WMMA accumulations ----
    {
      v16h A[4], Bm[4];
#pragma unroll
      for (int d = 0; d < 4; ++d) {
        const _Float16* row = g16 + (d * 16 + p) * CC;
        // A 16x32 f16: lane half h -> K in [8h,8h+7] then [16+8h,16+8h+7]
        v8h lo = *(const v8h*)(row + 8 * half);
        v8h hi = *(const v8h*)(row + 16 + 8 * half);
        A[d] = __builtin_shufflevector(lo, hi, 0, 1, 2, 3, 4, 5, 6, 7,
                                       8, 9, 10, 11, 12, 13, 14, 15);
        // B 32x16 f16: lane half h -> K in [8h,8h+7] then [16+8h,16+8h+7], N = p
        Bm[d] = A[d];
      }
#if defined(__HIP_DEVICE_COMPILE__)
      accH  = __builtin_amdgcn_wmma_f32_16x16x32_f16(false, A[0], false, Bm[1], (short)0, accH,  false, false);
      accV  = __builtin_amdgcn_wmma_f32_16x16x32_f16(false, A[2], false, Bm[3], (short)0, accV,  false, false);
      accNR = __builtin_amdgcn_wmma_f32_16x16x32_f16(false, A[0], false, Bm[0], (short)0, accNR, false, false);
      accNL = __builtin_amdgcn_wmma_f32_16x16x32_f16(false, A[1], false, Bm[1], (short)0, accNL, false, false);
      accNT = __builtin_amdgcn_wmma_f32_16x16x32_f16(false, A[2], false, Bm[2], (short)0, accNT, false, false);
      accNB = __builtin_amdgcn_wmma_f32_16x16x32_f16(false, A[3], false, Bm[3], (short)0, accNB, false, false);
#endif
    }
    __syncthreads();
  }

  // ---- diagonal extraction: (p,p) lives in lane p (VGPR p) for p<8 and
  //      lane p+16 (VGPR p-8) for p>=8, per the 16x16 f32 C/D layout. ----
  const bool hasDiag = (lane < 8) || (lane >= 24);
  const int vsel = (lane < 8) ? lane : ((lane >= 24) ? lane - 24 : 0);
  const int pout = (lane < 8) ? lane : lane - 16;
  float dH = pick8(accH, vsel);
  float dV = pick8(accV, vsel);
  float nR = pick8(accNR, vsel);
  float nL = pick8(accNL, vsel);
  float nT = pick8(accNT, vsel);
  float nB = pick8(accNB, vsel);
  float res = 0.5f * (dH * rsqrtf(nR * nL) + dV * rsqrtf(nT * nB));
  const int gx = x0 + pout;
  if (hasDiag && gx < WW) out[(size_t)b * (HH * WW) + (size_t)y * WW + gx] = res;
}

extern "C" void kernel_launch(void* const* d_in, const int* in_sizes, int n_in,
                              void* d_out, int out_size, void* d_ws, size_t ws_size,
                              hipStream_t stream) {
  (void)in_sizes; (void)n_in; (void)out_size; (void)d_ws; (void)ws_size;
  const float* features = (const float*)d_in[0];
  float* out = (float*)d_out;

  (void)hipFuncSetAttribute(
      (const void*)ComputeTotalSim_84267258347855_kernel,
      hipFuncAttributeMaxDynamicSharedMemorySize, (int)SMEM_BYTES);

  dim3 grid((WW + TPX - 1) / TPX, (HH + TPY - 1) / TPY, NB);  // 4 x 7 x 16
  ComputeTotalSim_84267258347855_kernel<<<grid, dim3(256), SMEM_BYTES, stream>>>(
      features, out);
}
